// RelGraphLayer_69647189672496
// MI455X (gfx1250) — matrix operand order repaired and checked by reading, not compiled, not run on hardware
//
#include <hip/hip_runtime.h>
#include <hip/hip_bf16.h>

// CDNA5 / gfx1250, wave32. WMMA fragment types per CDNA5_HIP.md.
typedef __attribute__((ext_vector_type(16))) __bf16 v16bf;
typedef __attribute__((ext_vector_type(8)))  __bf16 v8bf;
typedef __attribute__((ext_vector_type(8)))  float  v8f;

#define D 128   // D_IN == D_OUT

// ---------------------------------------------------------------------------
// Zero a float buffer (float4 granularity). Graph-capture safe.
// ---------------------------------------------------------------------------
__global__ __launch_bounds__(256) void zero_f32(float* __restrict__ p, int n4) {
  int i = blockIdx.x * 256 + threadIdx.x;
  if (i < n4) ((float4*)p)[i] = make_float4(0.f, 0.f, 0.f, 0.f);
}

// ---------------------------------------------------------------------------
// One-time weight prep: W [nMat,128,128] f32 row-major (k,n) ->
// Wt [nMat,128,128] bf16 transposed (n,k). 590 KB total; runs once per call.
// ---------------------------------------------------------------------------
__global__ __launch_bounds__(256) void convert_weights_bf16t(
    const float* __restrict__ W, __bf16* __restrict__ Wt, int total) {
  int i = blockIdx.x * 256 + threadIdx.x;
  if (i >= total) return;
  int mat = i >> 14;          // / (128*128)
  int rem = i & 16383;
  int n = rem >> 7;
  int k = rem & (D - 1);
  Wt[i] = (__bf16)W[(mat << 14) + (k << 7) + n];
}

// ---------------------------------------------------------------------------
// Edge scatter in D_IN space: S[dst] += ew * x[src].
// One wave per edge; each lane moves 4 of 128 channels (float4 load + 4 f32
// global atomics). x (25.6MB) and S (25.6MB) are L2-resident (192MB L2).
// ---------------------------------------------------------------------------
__global__ __launch_bounds__(256) void scatter_edges(
    const int*   __restrict__ src,
    const int*   __restrict__ dst,
    const float* __restrict__ ew,
    const float* __restrict__ x,
    float*       __restrict__ S,
    int nEdges) {
  int e = blockIdx.x * 8 + (threadIdx.x >> 5);   // wave-per-edge
  if (e >= nEdges) return;
  int lane = threadIdx.x & 31;
  int s = src[e];
  int d = dst[e];
  float w = ew[e];
  float4 v = ((const float4*)(x + (size_t)s * D))[lane];
  float* o = S + (size_t)d * D + lane * 4;
  atomicAdd(o + 0, w * v.x);
  atomicAdd(o + 1, w * v.y);
  atomicAdd(o + 2, w * v.z);
  atomicAdd(o + 3, w * v.w);
}

// ---------------------------------------------------------------------------
// Node-level GEMM: C(+)= A[M,128] @ B[128,128]; A f32, B pre-converted bf16
// transposed (n,k), C f32. Block = 16-row M tile, 8 waves; wave w owns N-tile
// [16w,16w+16). All of B^T staged in LDS (32KB) via coalesced float4 copies;
// A tile converted f32->bf16 during staging (4KB LDS). Inner loop per K-step:
// 2+2 ds_load_b128 + one v_wmma_f32_16x16x32_bf16; K=128 -> 4 WMMAs/wave.
// Bounds: tile origin is clamped (m0 = min(bid*16, M-16)) so the epilogue is
// branch-free; for this problem M=50000=3125*16 so tiles cover M exactly and
// no overlap occurs (required for the ACC read-modify-write variant).
// ---------------------------------------------------------------------------
template <bool ACC>
__global__ __launch_bounds__(256) void gemm_nodes(
    const float*  __restrict__ A,    // [M, 128] f32
    const __bf16* __restrict__ Bt,   // [128, 128] bf16, (n,k) layout
    float*        __restrict__ C,    // [M, 128] f32
    int M) {
  __shared__ __bf16 Bs[D * D];       // 32 KB, (n,k)
  __shared__ __bf16 As[16 * D];      // 4 KB,  (m,k)

  const int tid  = threadIdx.x;
  const int wave = tid >> 5;          // 0..7 -> N tile
  const int lane = tid & 31;
  const int half = lane >> 4;         // lane group 0-15 / 16-31
  const int l    = lane & 15;
  int m0 = blockIdx.x * 16;
  if (m0 > M - 16) m0 = M - 16;       // clamp: branch-free epilogue
  const int n0 = wave * 16;

  // Stage B^T: 16384 bf16 = 32 KB = 2048 float4, 8 per thread, coalesced.
  {
    const float4* g = (const float4*)Bt;
    float4*       s = (float4*)Bs;
#pragma unroll
    for (int i = 0; i < 8; ++i) s[tid + 256 * i] = g[tid + 256 * i];
  }
  // Stage A tile with in-flight f32->bf16: 2048 f32, 8 per thread.
  {
    int idx = tid * 8;
    int row = idx >> 7;               // / 128
    int col = idx & (D - 1);
    const float4* g = (const float4*)(A + (size_t)(m0 + row) * D + col);
    float4 v0 = g[0];
    float4 v1 = g[1];
    v8bf h;
    h[0] = (__bf16)v0.x; h[1] = (__bf16)v0.y;
    h[2] = (__bf16)v0.z; h[3] = (__bf16)v0.w;
    h[4] = (__bf16)v1.x; h[5] = (__bf16)v1.y;
    h[6] = (__bf16)v1.z; h[7] = (__bf16)v1.w;
    *(v8bf*)(As + row * D + col) = h;
  }
  if (ACC) {
    // Warm L2/L0 for the C rows we will read-modify-write.
    __builtin_prefetch(C + (size_t)(m0 + (tid >> 4)) * D, 0, 3);
  }
  __syncthreads();

  // C/D layout (32-bit 16x16): VGPR v -> row v (lanes 0-15) / v+8 (16-31),
  // col = lane & 15.
  float* __restrict__ cbase = C + (size_t)(m0 + 8 * half) * D + n0 + l;
  v8f acc;
#pragma unroll
  for (int v = 0; v < 8; ++v)
    acc[v] = ACC ? cbase[(size_t)v * D] : 0.f;

#pragma unroll
  for (int k0 = 0; k0 < D; k0 += 32) {
    // A fragment (16-bit A 16x32): lane row m=l; elements 0-7 at K=k0+8h..+7,
    // elements 8-15 at K=k0+16+8h..+7. Two 16B LDS loads.
    const v8bf* ap = (const v8bf*)(As + l * D + k0 + 8 * half);
    v8bf a0 = ap[0];
    v8bf a1 = ap[2];                  // +16 bf16
    v16bf a = __builtin_shufflevector(a0, a1, 0, 1, 2, 3, 4, 5, 6, 7,
                                      8, 9, 10, 11, 12, 13, 14, 15);
    // B fragment (32x16): col n0+l; elements 0-7 at K=k0+8h..+7, elements
    // 8-15 at K=k0+16+8h..+7 (same as A). (n,k) LDS layout -> two 16B loads.
    const v8bf* bp = (const v8bf*)(Bs + (n0 + l) * D + k0 + 8 * half);
    v8bf b0 = bp[0];
    v8bf b1 = bp[2];
    v16bf b = __builtin_shufflevector(b0, b1, 0, 1, 2, 3, 4, 5, 6, 7,
                                      8, 9, 10, 11, 12, 13, 14, 15);

    acc = __builtin_amdgcn_wmma_f32_16x16x32_bf16(
        /*neg_a=*/false, a, /*neg_b=*/false, b,
        /*c_mod=*/(short)0, acc, /*reuse_a=*/false, /*reuse_b=*/false);
  }

#pragma unroll
  for (int v = 0; v < 8; ++v)
    cbase[(size_t)v * D] = acc[v];
}

// ---------------------------------------------------------------------------
// Final epilogue: out = relu(out + root_b + bias).
// ---------------------------------------------------------------------------
__global__ __launch_bounds__(256) void bias_relu(
    float* __restrict__ out, const float* __restrict__ rb,
    const float* __restrict__ bias, int n) {
  int i = blockIdx.x * 256 + threadIdx.x;
  if (i >= n) return;
  int c = i & (D - 1);
  float v = out[i] + rb[c] + bias[c];
  out[i] = v > 0.f ? v : 0.f;
}

// ---------------------------------------------------------------------------
// Launch: weight prep, root GEMM, then per relation {zero S, scatter edges,
// GEMM-accum}, then bias+relu. All stream-ordered; RMW of d_out is safe under
// stream serialization. Workspace: 25.6 MB scatter buffer + 294 KB bf16 W^T.
// ---------------------------------------------------------------------------
extern "C" void kernel_launch(void* const* d_in, const int* in_sizes, int n_in,
                              void* d_out, int out_size, void* d_ws, size_t ws_size,
                              hipStream_t stream) {
  const float* x    = (const float*)d_in[0];
  const int*   src  = (const int*)  d_in[1];
  const int*   dst  = (const int*)  d_in[2];
  const float* ew   = (const float*)d_in[3];
  const float* w    = (const float*)d_in[4];   // [R, 128, 128]
  const float* rw   = (const float*)d_in[5];   // [128, 128]
  const float* rb   = (const float*)d_in[6];
  const float* bias = (const float*)d_in[7];
  float* out = (float*)d_out;

  const int nNodes = in_sizes[0] / D;          // 50000
  const int nRel   = in_sizes[4] / (D * D);    // 8
  const int nEdge  = in_sizes[1] / nRel;       // 64000

  float*  S  = (float*)d_ws;                   // [nNodes, 128] f32 scratch
  __bf16* Wt = (__bf16*)((char*)d_ws + (size_t)nNodes * D * sizeof(float));

  const int tilesM  = (nNodes + 15) / 16;            // 3125 (exact cover)
  const int zblocks = (nNodes * (D / 4) + 255) / 256;
  const int sblocks = (nEdge + 7) / 8;

  // One-time bf16 transposed weight prep: root first, then the 8 relations.
  convert_weights_bf16t<<<(D * D + 255) / 256, 256, 0, stream>>>(rw, Wt, D * D);
  convert_weights_bf16t<<<(nRel * D * D + 255) / 256, 256, 0, stream>>>(
      w, Wt + D * D, nRel * D * D);

  // out = x @ root_w
  gemm_nodes<false><<<tilesM, 256, 0, stream>>>(x, Wt, out, nNodes);

  for (int r = 0; r < nRel; ++r) {
    zero_f32<<<zblocks, 256, 0, stream>>>(S, nNodes * (D / 4));
    scatter_edges<<<sblocks, 256, 0, stream>>>(
        src + (size_t)r * nEdge, dst + (size_t)r * nEdge,
        ew + (size_t)r * nEdge, x, S, nEdge);
    // out += S @ W_r
    gemm_nodes<true><<<tilesM, 256, 0, stream>>>(
        S, Wt + (size_t)(r + 1) * D * D, out, nNodes);
  }

  bias_relu<<<(out_size + 255) / 256, 256, 0, stream>>>(out, rb, bias, out_size);
}
